// TGCNN_layer_7610682049107
// MI455X (gfx1250) — compile-verified, matrix operand order checked by reading, not run on hardware
//
#include <hip/hip_runtime.h>

// Problem constants (from the reference)
#define BATCH   32
#define NPAIRS  10000      // N*N
#define TT      100        // T
#define FS      3
#define FDIM    16
#define NWIN    98         // (T - FS)/STRIDE + 1

// Tiling
#define PB      128        // pairs per block (K-chunk)
#define PBPAD   136        // LDS row pitch in elements (272B rows: 16B aligned, bank-spread)
#define TROWS   114        // 7*16 window rows + FS-1 lookahead, zero padded past t=99
#define NWAVES  7          // one wave per 16-window M tile (7*16 = 112 >= 98)
#define BLOCK   (NWAVES * 32)

typedef __bf16 bf16x16 __attribute__((ext_vector_type(16)));
typedef float  f32x8   __attribute__((ext_vector_type(8)));
typedef float  f32x4   __attribute__((ext_vector_type(4)));

union Frag {
    bf16x16 bf;
    f32x4   f4[2];
};

__global__ void tgcnn_zero_kernel(float* __restrict__ out, int n) {
    int i = blockIdx.x * blockDim.x + threadIdx.x;
    if (i < n) out[i] = 0.0f;
}

__global__ __launch_bounds__(BLOCK)
void tgcnn_wmma_kernel(const float* __restrict__ x,      // (B, N, N, T) f32
                       const float* __restrict__ w,      // (N*N*FS, F)  f32
                       const float* __restrict__ gammat, // scalar
                       float* __restrict__ out)          // (B, F, 1, NWIN) f32, pre-zeroed
{
    __shared__ __align__(16) __bf16 xeT[TROWS * PBPAD];   // xeT[t][pair]  (A staged, t-major)
    __shared__ __align__(16) __bf16 wbT[FS * FDIM * PBPAD]; // wbT[fs*16+f][pair] (B staged, f-major)

    const int b   = blockIdx.x;
    const int p0  = blockIdx.y * PB;   // first global (i,j) pair of this chunk
    const int tid = threadIdx.x;
    const float g = gammat[0];

    // ---- Phase 1: stream x chunk from HBM, exp-transform, stage transposed bf16 in LDS
    const float* xb = x + (size_t)b * NPAIRS * TT;
    for (int e = tid; e < PB * TT; e += BLOCK) {
        const int p = e / TT;          // local pair
        const int t = e % TT;          // time (contiguous in global -> coalesced)
        const int pg = p0 + p;
        float v = (pg < NPAIRS) ? xb[(size_t)pg * TT + t] : 0.0f;
        float r = (v != 0.0f) ? __expf(-g * v) : 0.0f;   // sparse map_values(exp(-gamma*v))
        xeT[t * PBPAD + p] = (__bf16)r;
    }
    // zero the lookahead rows t in [100, 114) so the last M tile contracts zeros
    for (int e = tid; e < (TROWS - TT) * PB; e += BLOCK) {
        const int t = TT + e / PB;
        const int p = e % PB;
        xeT[t * PBPAD + p] = (__bf16)0.0f;
    }

    // ---- Phase 2: stage this chunk's weights (L2-resident), bf16, f-major rows
    for (int e = tid; e < PB * (FS * FDIM); e += BLOCK) {
        const int p = e / (FS * FDIM);
        const int r = e % (FS * FDIM);          // r = fs*16 + f ; w is (pair*48 + r) contiguous
        const int pg = p0 + p;
        float v = (pg < NPAIRS) ? w[(size_t)pg * (FS * FDIM) + r] : 0.0f;
        wbT[r * PBPAD + p] = (__bf16)v;
    }

    __syncthreads();

    // ---- Phase 3: WMMA. D[win, f] += A[win, pair] * B[pair, f] for 3 shifted views (fs)
    const int wave = tid >> 5;
    const int lane = tid & 31;
    if (wave < NWAVES) {
        const int winbase = wave * 16;
        const int mn    = lane & 15;               // A row (win offset) == B/D column (f)
        const int khalf = (lane & 16) ? 8 : 0;     // lanes 0-15: K 0..7,16..23 ; 16-31: 8..15,24..31
        f32x8 acc = {0.f, 0.f, 0.f, 0.f, 0.f, 0.f, 0.f, 0.f};

        for (int fs = 0; fs < FS; ++fs) {
            const __bf16* arow = &xeT[(winbase + mn + fs) * PBPAD]; // A row m: xe[.][win+fs]
            const __bf16* brow = &wbT[(fs * FDIM + mn) * PBPAD];    // B col f (f-major rows)
            #pragma unroll
            for (int kt = 0; kt < PB; kt += 32) {
                Frag a, bb;
                a.f4[0]  = *(const f32x4*)(arow + kt + khalf);        // K + 0..7  (packed pairs)
                a.f4[1]  = *(const f32x4*)(arow + kt + khalf + 16);   // K + 16..23
                bb.f4[0] = *(const f32x4*)(brow + kt + khalf);
                bb.f4[1] = *(const f32x4*)(brow + kt + khalf + 16);
                acc = __builtin_amdgcn_wmma_f32_16x16x32_bf16(
                        /*neg_a=*/false, a.bf, /*neg_b=*/false, bb.bf,
                        /*c_mod=*/(short)0, acc, /*reuse_a=*/false, /*reuse_b=*/false);
            }
        }

        // C/D layout: VGPR r -> M = r (lanes 0-15) or r+8 (lanes 16-31), N = lane&15
        float* ob = out + ((size_t)b * FDIM + mn) * NWIN;  // out[b][f][0][win]
        #pragma unroll
        for (int r2 = 0; r2 < 8; ++r2) {
            const int win = winbase + r2 + ((lane & 16) ? 8 : 0);
            if (win < NWIN) atomicAdd(&ob[win], acc[r2]);  // accumulate K-chunk partials
        }
    }
}

extern "C" void kernel_launch(void* const* d_in, const int* in_sizes, int n_in,
                              void* d_out, int out_size, void* d_ws, size_t ws_size,
                              hipStream_t stream) {
    const float* x      = (const float*)d_in[0];   // (32, 100, 100, 100) f32
    const float* w      = (const float*)d_in[1];   // (30000, 16) f32
    const float* gammat = (const float*)d_in[2];   // scalar
    float* out          = (float*)d_out;           // (32, 16, 1, 98) f32

    tgcnn_zero_kernel<<<(out_size + 255) / 256, 256, 0, stream>>>(out, out_size);

    dim3 grid(BATCH, (NPAIRS + PB - 1) / PB);      // 32 x 79 blocks
    tgcnn_wmma_kernel<<<grid, BLOCK, 0, stream>>>(x, w, gammat, out);
}
